// HilbertAttentionCoreFixed_69776038690878
// MI455X (gfx1250) — matrix operand order repaired and checked by reading, not compiled, not run on hardware
//
#include <hip/hip_runtime.h>
#include <stdint.h>

typedef __bf16 bf16_t;
typedef __attribute__((ext_vector_type(16))) __bf16 v16bf;
typedef __attribute__((ext_vector_type(8)))  __bf16 v8bf;
typedef __attribute__((ext_vector_type(8)))  float  v8f;

#define NH   16
#define SEG  2048
#define DIL  8
#define BM   64
#define HD   64
#define BB   2
#define MM   8192
#define DD   1024
#define DD3  3072
#define NSEG (MM / SEG)          // 4
#define KN   (SEG / DIL)         // 256 dilated keys per segment
#define NB   (SEG / BM)          // 32 query blocks per segment
#define QBLOCKS (BB * NH * NSEG * NB)  // 4096
#define LSTR 40                  // LDS row stride (elems): 80B = 16B multiple, bank-staggered

// Native f32->bf16 conversion (gfx1250 has bf16 cvt hardware; single VALU op,
// pairable into v_cvt_pk_bf16_f32 by the backend).
__device__ __forceinline__ bf16_t f2bf(float f) {
  return (bf16_t)f;
}

__device__ __forceinline__ v8f wmma_bf16(v16bf a, v16bf b, v8f c) {
  // (neg_a, A, neg_b, B, c_mod, C, reuse_a, reuse_b) -> v_wmma_f32_16x16x32_bf16
  return __builtin_amdgcn_wmma_f32_16x16x32_bf16(false, a, false, b, (short)0, c, false, false);
}

__device__ __forceinline__ v16bf cat8(v8bf lo, v8bf hi) {
  return __builtin_shufflevector(lo, hi, 0,1,2,3,4,5,6,7,8,9,10,11,12,13,14,15);
}

__device__ __forceinline__ v8f vzero8() {
  return (v8f){0.f,0.f,0.f,0.f,0.f,0.f,0.f,0.f};
}

__global__ void cvt_f32_bf16(const float* __restrict__ in, bf16_t* __restrict__ out, int n) {
  int i = blockIdx.x * blockDim.x + threadIdx.x;
  if (i < n) out[i] = f2bf(in[i]);
}

// C[m,n] = sum_k X[m,k] * W[n,k]   (torch Linear: x @ W^T)
// QKV variant: K=1024, N=3072; scale q columns (n<1024) by hd^-0.5; bf16 out, row stride 3072.
__global__ __launch_bounds__(128) void gemm_qkv(const bf16_t* __restrict__ X,
                                                const bf16_t* __restrict__ W,
                                                bf16_t* __restrict__ C) {
  const int lane = threadIdx.x & 31;
  const int half = lane >> 4, l16 = lane & 15;
  const int mt = blockIdx.x * 4 + (threadIdx.x >> 5);
  const int n0 = blockIdx.y * 64;
  const int m0 = mt * 16;
  const bf16_t* xrow = X + (size_t)(m0 + l16) * DD;

  v8f acc[4];
#pragma unroll
  for (int i = 0; i < 4; i++) acc[i] = vzero8();

  for (int k0 = 0; k0 < DD; k0 += 32) {
    // A frag: 16x32, lane m=l16, K chunk half*16 (contiguous 32B, two 16B loads)
    v16bf a = cat8(*(const v8bf*)(xrow + k0 + half*16),
                   *(const v8bf*)(xrow + k0 + half*16 + 8));
#pragma unroll
    for (int nt = 0; nt < 4; nt++) {
      // B frag: 32x16, lane n=l16, K chunk half*16 (contiguous 32B)
      const bf16_t* wrow = W + (size_t)(n0 + nt*16 + l16) * DD + k0 + half*16;
      v16bf bfr = cat8(*(const v8bf*)wrow, *(const v8bf*)(wrow + 8));
      acc[nt] = wmma_bf16(a, bfr, acc[nt]);
    }
  }
#pragma unroll
  for (int nt = 0; nt < 4; nt++) {
    int n = n0 + nt*16 + l16;
    float sc = (n < DD) ? 0.125f : 1.0f;   // q pre-scaled by 64^-0.5
    bf16_t* crow = C + (size_t)(m0 + 8*half) * DD3 + n;
#pragma unroll
    for (int r = 0; r < 8; r++) crow[(size_t)r * DD3] = f2bf(acc[nt][r] * sc);
  }
}

// Out projection: K=1024, N=1024, fp32 output row stride 1024.
__global__ __launch_bounds__(128) void gemm_out(const bf16_t* __restrict__ X,
                                                const bf16_t* __restrict__ W,
                                                float* __restrict__ C) {
  const int lane = threadIdx.x & 31;
  const int half = lane >> 4, l16 = lane & 15;
  const int mt = blockIdx.x * 4 + (threadIdx.x >> 5);
  const int n0 = blockIdx.y * 64;
  const int m0 = mt * 16;
  const bf16_t* xrow = X + (size_t)(m0 + l16) * DD;

  v8f acc[4];
#pragma unroll
  for (int i = 0; i < 4; i++) acc[i] = vzero8();

  for (int k0 = 0; k0 < DD; k0 += 32) {
    v16bf a = cat8(*(const v8bf*)(xrow + k0 + half*16),
                   *(const v8bf*)(xrow + k0 + half*16 + 8));
#pragma unroll
    for (int nt = 0; nt < 4; nt++) {
      const bf16_t* wrow = W + (size_t)(n0 + nt*16 + l16) * DD + k0 + half*16;
      v16bf bfr = cat8(*(const v8bf*)wrow, *(const v8bf*)(wrow + 8));
      acc[nt] = wmma_bf16(a, bfr, acc[nt]);
    }
  }
#pragma unroll
  for (int nt = 0; nt < 4; nt++) {
    float* crow = C + (size_t)(m0 + 8*half) * DD + n0 + nt*16 + l16;
#pragma unroll
    for (int r = 0; r < 8; r++) crow[(size_t)r * DD] = acc[nt][r];
  }
}

// Dilated-segment attention. One wave handles one 64-query block of one (b,h,seg).
// Streams dilated keys in chunks of 32; per-key max over the query block (reference
// normalization) needs no running rescale, so num/den accumulate directly.
__global__ __launch_bounds__(128) void dilated_attn(const bf16_t* __restrict__ QKV,
                                                    bf16_t* __restrict__ OUT) {
  __shared__ bf16_t lds[4][(HD + BM) * LSTR];    // per-wave: V^T (64xLSTR) + W (64xLSTR)
  const int lane = threadIdx.x & 31;
  const int wv = threadIdx.x >> 5;
  const int half = lane >> 4, l16 = lane & 15;
  bf16_t* VT = &lds[wv][0];              // [hd][key-in-chunk]
  bf16_t* WB = &lds[wv][HD * LSTR];      // [query][key-in-chunk]

  int qb = blockIdx.x * 4 + wv;
  int blk = qb % NB; int t = qb / NB;
  int s = t % NSEG;  t /= NSEG;
  int h = t % NH;    int b = t / NH;

  const size_t rowB = (size_t)b * MM;
  const int m0 = s * SEG + blk * BM;

  // Q fragments (pre-scaled), 4 M-tiles x 2 k-steps over hd=64
  v16bf qf[4][2];
#pragma unroll
  for (int mt = 0; mt < 4; mt++) {
    const bf16_t* qrow = QKV + (rowB + m0 + mt*16 + l16) * (size_t)DD3 + h * HD;
#pragma unroll
    for (int ks = 0; ks < 2; ks++)
      qf[mt][ks] = cat8(*(const v8bf*)(qrow + ks*32 + half*16),
                        *(const v8bf*)(qrow + ks*32 + half*16 + 8));
  }

  const bf16_t* Kb = QKV + (rowB + s * SEG) * (size_t)DD3 + DD   + h * HD;
  const bf16_t* Vb = QKV + (rowB + s * SEG) * (size_t)DD3 + 2*DD + h * HD;
  const size_t kstride = (size_t)DIL * DD3;   // dilated key row stride

  v8f num[4][4];
  v8f den[4];
#pragma unroll
  for (int mt = 0; mt < 4; mt++) {
    den[mt] = vzero8();
#pragma unroll
    for (int nt = 0; nt < 4; nt++) num[mt][nt] = vzero8();
  }

  for (int j0 = 0; j0 < KN; j0 += 32) {
    // Stage V^T: lane owns dilated key (j0+lane); scatter its 64 hd values column-wise.
    const bf16_t* vrow = Vb + (size_t)(j0 + lane) * kstride;
#pragma unroll
    for (int hh = 0; hh < HD; hh += 8) {
      v8bf vv = *(const v8bf*)(vrow + hh);
#pragma unroll
      for (int e = 0; e < 8; e++) VT[(hh + e) * LSTR + lane] = vv[e];
    }

    // Scores S = Q K^T  : [64q x 32keys], hd=64 -> 2 k-steps
    v8f scr[4][2];
#pragma unroll
    for (int mt = 0; mt < 4; mt++) { scr[mt][0] = vzero8(); scr[mt][1] = vzero8(); }
#pragma unroll
    for (int nt = 0; nt < 2; nt++) {
      const bf16_t* krow = Kb + (size_t)(j0 + nt*16 + l16) * kstride + half*16;
      v16bf b0 = cat8(*(const v8bf*)(krow),      *(const v8bf*)(krow + 8));
      v16bf b1 = cat8(*(const v8bf*)(krow + 32), *(const v8bf*)(krow + 40));
#pragma unroll
      for (int mt = 0; mt < 4; mt++) {
        scr[mt][nt] = wmma_bf16(qf[mt][0], b0, scr[mt][nt]);
        scr[mt][nt] = wmma_bf16(qf[mt][1], b1, scr[mt][nt]);
      }
    }

    // Per-key max over the 64-query block, exp, den accumulation, stage W.
#pragma unroll
    for (int nt = 0; nt < 2; nt++) {
      float cm = -3.402823466e38f;
#pragma unroll
      for (int mt = 0; mt < 4; mt++)
#pragma unroll
        for (int r = 0; r < 8; r++) cm = fmaxf(cm, scr[mt][nt][r]);
      cm = fmaxf(cm, __shfl_xor(cm, 16, 32));   // lanes L and L^16 share key column
#pragma unroll
      for (int mt = 0; mt < 4; mt++) {
#pragma unroll
        for (int r = 0; r < 8; r++) {
          float w = __expf(scr[mt][nt][r] - cm);
          scr[mt][nt][r] = w;
          den[mt][r] += w;
        }
#pragma unroll
        for (int r = 0; r < 8; r++)
          WB[(mt*16 + r + 8*half) * LSTR + nt*16 + l16] = f2bf(scr[mt][nt][r]);
      }
    }

    // num += W V : A = W (16x32 keys), B from V^T (contiguous 16B LDS reads).
    // DS ops are in-order within a wave, so the re-layout round trip is coherent.
#pragma unroll
    for (int mt = 0; mt < 4; mt++) {
      const bf16_t* wrow = WB + (mt*16 + l16) * LSTR;
      v16bf af = cat8(*(const v8bf*)(wrow + half*16),
                      *(const v8bf*)(wrow + half*16 + 8));
#pragma unroll
      for (int nt = 0; nt < 4; nt++) {
        const bf16_t* vtr = VT + (nt*16 + l16) * LSTR + half*16;
        v16bf bf_ = cat8(*(const v8bf*)vtr, *(const v8bf*)(vtr + 8));
        num[mt][nt] = wmma_bf16(af, bf_, num[mt][nt]);
      }
    }
  }

  // Reduce den across key lanes (xor 1..8 stays within each 16-lane half,
  // which is exactly the half owning rows m = r + 8*half).
#pragma unroll
  for (int mt = 0; mt < 4; mt++) {
#pragma unroll
    for (int r = 0; r < 8; r++) {
      float d = den[mt][r];
      d += __shfl_xor(d, 1, 32);
      d += __shfl_xor(d, 2, 32);
      d += __shfl_xor(d, 4, 32);
      d += __shfl_xor(d, 8, 32);
      den[mt][r] = d + 1e-10f;
    }
  }

  // out = num / den, bf16 into [B*M, D] at columns h*HD + ...
#pragma unroll
  for (int mt = 0; mt < 4; mt++) {
#pragma unroll
    for (int nt = 0; nt < 4; nt++) {
      bf16_t* orow = OUT + (rowB + m0 + mt*16 + 8*half) * (size_t)DD + h*HD + nt*16 + l16;
#pragma unroll
      for (int r = 0; r < 8; r++)
        orow[(size_t)r * DD] = f2bf(num[mt][nt][r] / den[mt][r]);
    }
  }
}

extern "C" void kernel_launch(void* const* d_in, const int* in_sizes, int n_in,
                              void* d_out, int out_size, void* d_ws, size_t ws_size,
                              hipStream_t stream) {
  const float* x    = (const float*)d_in[0];
  const float* Wqkv = (const float*)d_in[1];
  const float* Wout = (const float*)d_in[2];
  float* out = (float*)d_out;

  // Workspace layout (bf16):
  //   xb[BB*MM*DD] | wqkvb[DD3*DD] | woutb[DD*DD] | qkv[BB*MM*DD3] | attn[BB*MM*DD]
  bf16_t* xb    = (bf16_t*)d_ws;
  bf16_t* wqkvb = xb    + (size_t)BB*MM*DD;
  bf16_t* woutb = wqkvb + (size_t)DD3*DD;
  bf16_t* qkv   = woutb + (size_t)DD*DD;
  bf16_t* attn  = qkv   + (size_t)BB*MM*DD3;

  int nx = BB*MM*DD;
  cvt_f32_bf16<<<(nx + 255)/256, 256, 0, stream>>>(x, xb, nx);
  cvt_f32_bf16<<<(DD3*DD + 255)/256, 256, 0, stream>>>(Wqkv, wqkvb, DD3*DD);
  cvt_f32_bf16<<<(DD*DD + 255)/256, 256, 0, stream>>>(Wout, woutb, DD*DD);

  gemm_qkv<<<dim3((BB*MM/16)/4, DD3/64), 128, 0, stream>>>(xb, wqkvb, qkv);
  dilated_attn<<<QBLOCKS/4, 128, 0, stream>>>(qkv, attn);
  gemm_out<<<dim3((BB*MM/16)/4, DD/64), 128, 0, stream>>>(attn, woutb, out);
}
